// Discriminator_69088843924193
// MI455X (gfx1250) — matrix operand order repaired and checked by reading, not compiled, not run on hardware
//
#include <hip/hip_runtime.h>
#include <hip/hip_bf16.h>

#define C 256
#define N_ATOMS 16384
#define LAYERS 4
#define SLOPE 0.2f
#define RING 512
#define LDS_STRIDE 264   // padded row stride (elements) for bank-conflict avoidance

typedef __attribute__((ext_vector_type(8)))  unsigned short v8u;
typedef __attribute__((ext_vector_type(16))) __bf16         v16bf;
typedef __attribute__((ext_vector_type(8)))  float          v8f;

union Frag16 {
    v16bf bf;
    struct { v8u lo, hi; } u;
};

__device__ __forceinline__ unsigned short f2bf(float f) {
    unsigned u = __builtin_bit_cast(unsigned, f);
    // round-to-nearest-even bf16
    unsigned r = (u + 0x7FFFu + ((u >> 16) & 1u)) >> 16;
    return (unsigned short)r;
}

__device__ __forceinline__ float lrelu(float v) {
    return v >= 0.f ? v : SLOPE * v;
}

// ---------------------------------------------------------------------------
// Length stack: (1,1) -> (1,C), two lrelu layers + linear.  One workgroup.
// ---------------------------------------------------------------------------
__global__ void k_len(const float* l,
                      const float* w0, const float* b0,
                      const float* w1, const float* b1,
                      const float* w2, const float* b2,
                      float* lv)
{
    __shared__ float t0[C], t1[C];
    int c = threadIdx.x;
    float lval = l[0];
    float v = lrelu(lval * w0[c] + b0[c]);
    t0[c] = v; __syncthreads();
    float a = b1[c];
    #pragma unroll 8
    for (int k = 0; k < C; k++) a += t0[k] * w1[k * C + c];
    t1[c] = lrelu(a); __syncthreads();
    float o = b2[c];
    #pragma unroll 8
    for (int k = 0; k < C; k++) o += t1[k] * w2[k * C + c];
    lv[c] = o;
}

// ---------------------------------------------------------------------------
// First MLP layer: (N,10) @ (10,C) + b, lrelu, output bf16.
// One block per atom; K=10 is too small for WMMA, VALU is optimal.
// ---------------------------------------------------------------------------
__global__ void k_in0(const float* __restrict__ x, const float* __restrict__ w0,
                      const float* __restrict__ b0, unsigned short* __restrict__ y0)
{
    __shared__ float xs[10];
    int n = blockIdx.x, c = threadIdx.x;
    if (c < 10) xs[c] = x[n * 10 + c];
    __syncthreads();
    float acc = b0[c];
    #pragma unroll
    for (int k = 0; k < 10; k++) acc += xs[k] * w0[k * C + c];
    y0[n * C + c] = f2bf(lrelu(acc));
}

// ---------------------------------------------------------------------------
// Transpose + f32->bf16 convert a CxC weight: WT[n][k] = bf16(W[k][n]).
// Makes B-fragment loads contiguous per lane.
// ---------------------------------------------------------------------------
__global__ void k_twt(const float* __restrict__ W, unsigned short* __restrict__ WT)
{
    int n = blockIdx.x, k = threadIdx.x;
    WT[n * C + k] = f2bf(W[k * C + n]);
}

// ---------------------------------------------------------------------------
// bf16 WMMA GEMM: out = act(A(N x 256, bf16) @ W(256x256) + bias)
// Wt (pre-transposed bf16) is staged in LDS (padded stride).  Each block:
// 8 waves x 16 rows = 128 rows; each wave sweeps all 16 column tiles,
// 8 x v_wmma_f32_16x16x32_bf16 per tile (K=256).
// ACT: 0=none, 1=lrelu.  OUTF32: write f32 (for P0) instead of bf16.
// ---------------------------------------------------------------------------
template <int ACT, int OUTF32>
__global__ void k_gemm(const unsigned short* __restrict__ A,
                       const unsigned short* __restrict__ WT,
                       const float* __restrict__ bias,
                       unsigned short* __restrict__ outB,
                       float* __restrict__ outF)
{
    extern __shared__ unsigned short sW[];   // 256 rows x LDS_STRIDE bf16
    const int tid = threadIdx.x;

    // Stage Wt into LDS: thread t copies row t (512B) as 32 x b128.
    {
        const v8u* src = (const v8u*)(WT + tid * C);
        v8u* dst = (v8u*)(sW + tid * LDS_STRIDE);
        #pragma unroll
        for (int i = 0; i < 32; i++) dst[i] = src[i];
    }
    __syncthreads();

    const int wave    = tid >> 5;
    const int lane    = tid & 31;
    const int m       = lane & 15;       // row within tile (A) / col within tile (B,D)
    const int hiHalf  = lane >> 4;
    const int rowBase = blockIdx.x * 128 + wave * 16;

    // Preload all 8 A fragments for this wave's 16 rows (K = 0..255).
    Frag16 a[8];
    const unsigned short* arow = A + (size_t)(rowBase + m) * C;
    #pragma unroll
    for (int kc = 0; kc < 8; kc++) {
        int kb = kc * 32 + hiHalf * 8;
        a[kc].u.lo = *(const v8u*)(arow + kb);
        a[kc].u.hi = *(const v8u*)(arow + kb + 16);
    }

    for (int nt = 0; nt < 16; nt++) {
        const int col = nt * 16 + m;
        v8f acc = {};
        const unsigned short* bcol = sW + col * LDS_STRIDE + hiHalf * 8;
        #pragma unroll
        for (int kc = 0; kc < 8; kc++) {
            Frag16 b;
            const unsigned short* bp = bcol + kc * 32;
            b.u.lo = *(const v8u*)bp;
            b.u.hi = *(const v8u*)(bp + 16);
            acc = __builtin_amdgcn_wmma_f32_16x16x32_bf16(
                false, a[kc].bf, false, b.bf, (short)0, acc, false, false);
        }
        const float bv = bias[col];
        #pragma unroll
        for (int i = 0; i < 8; i++) {
            float v = acc[i] + bv;
            if (ACT == 1) v = lrelu(v);
            const size_t r = (size_t)(rowBase + hiHalf * 8 + i);
            if (OUTF32) outF[r * C + col] = v;
            else        outB[r * C + col] = f2bf(v);
        }
    }
}

// ---------------------------------------------------------------------------
// 4-layer ReLU RNN scan, batch=1, diagonally pipelined: block l = layer l.
// Layer-0 input projection is precomputed (P0).  Cross-layer handoff through
// a global ring buffer with agent-scope release/acquire progress counters.
// Weight reads W[k*256+c] are perfectly coalesced and L2-resident (2 MB).
// ---------------------------------------------------------------------------
__global__ void k_rnn(const float* __restrict__ P0,
                      const float* __restrict__ wih, const float* __restrict__ whh,
                      const float* __restrict__ bih, const float* __restrict__ bhh,
                      float* __restrict__ Hring, float* __restrict__ hid,
                      int* __restrict__ prod, int nsteps)
{
    const int l = blockIdx.x;
    const int c = threadIdx.x;
    __shared__ float inv[C];
    __shared__ float hv[C];
    hv[c] = 0.f;
    __syncthreads();

    const float* Wih = wih + (size_t)l * C * C;
    const float* Whh = whh + (size_t)l * C * C;
    const float bsum = (l == 0) ? bhh[c] : (bih[l * C + c] + bhh[l * C + c]);
    float* myring = Hring + (size_t)l * RING * C;
    const float* inring = Hring + (size_t)(l - 1) * RING * C;

    for (int t = 0; t < nsteps; t++) {
        if (c == 0) {
            if (l > 0) {  // wait for h_{l-1}[t]
                while (__hip_atomic_load(&prod[l - 1], __ATOMIC_ACQUIRE,
                                         __HIP_MEMORY_SCOPE_AGENT) < t + 1)
                    __builtin_amdgcn_s_sleep(1);
            }
            if (l < 3 && t >= RING) {  // ring flow control vs consumer l+1
                while (__hip_atomic_load(&prod[l + 1], __ATOMIC_ACQUIRE,
                                         __HIP_MEMORY_SCOPE_AGENT) < t - RING + 1)
                    __builtin_amdgcn_s_sleep(1);
            }
        }
        __syncthreads();
        if (l > 0) inv[c] = inring[(size_t)(t & (RING - 1)) * C + c];
        __syncthreads();

        float acc = bsum;
        if (l == 0) {
            acc += P0[(size_t)t * C + c];
        } else {
            #pragma unroll 8
            for (int k = 0; k < C; k++) acc += inv[k] * Wih[k * C + c];
        }
        #pragma unroll 8
        for (int k = 0; k < C; k++) acc += hv[k] * Whh[k * C + c];
        acc = acc > 0.f ? acc : 0.f;

        __syncthreads();           // everyone done reading hv
        hv[c] = acc;
        if (l < 3) myring[(size_t)(t & (RING - 1)) * C + c] = acc;
        __syncthreads();
        if (c == 0 && l < 3)
            __hip_atomic_store(&prod[l], t + 1, __ATOMIC_RELEASE,
                               __HIP_MEMORY_SCOPE_AGENT);
    }
    hid[l * C + c] = hv[c];
}

// ---------------------------------------------------------------------------
// Final stack: feat(1,5C) -> C -> C -> C -> 1.  One workgroup.
// ---------------------------------------------------------------------------
__global__ void k_fin(const float* lv, const float* hid,
                      const float* w0, const float* b0,
                      const float* w1, const float* b1,
                      const float* w2, const float* b2,
                      const float* w3, const float* b3,
                      float* out)
{
    __shared__ float feat[5 * C];
    __shared__ float z0[C], z1[C], red[C];
    int c = threadIdx.x;
    feat[c] = lv[c];
    #pragma unroll
    for (int i = 0; i < 4; i++) feat[C + i * C + c] = hid[i * C + c];
    __syncthreads();
    float a = b0[c];
    #pragma unroll 8
    for (int k = 0; k < 5 * C; k++) a += feat[k] * w0[k * C + c];
    z0[c] = lrelu(a); __syncthreads();
    float d = b1[c];
    #pragma unroll 8
    for (int k = 0; k < C; k++) d += z0[k] * w1[k * C + c];
    z1[c] = lrelu(d); __syncthreads();
    float e = b2[c];
    #pragma unroll 8
    for (int k = 0; k < C; k++) e += z1[k] * w2[k * C + c];
    e = lrelu(e);
    red[c] = e * w3[c]; __syncthreads();
    for (int s = 128; s > 0; s >>= 1) {
        if (c < s) red[c] += red[c + s];
        __syncthreads();
    }
    if (c == 0) out[0] = red[0] + b3[0];
}

// ---------------------------------------------------------------------------
extern "C" void kernel_launch(void* const* d_in, const int* in_sizes, int n_in,
                              void* d_out, int out_size, void* d_ws, size_t ws_size,
                              hipStream_t stream)
{
    const float* x       = (const float*)d_in[0];
    const float* l       = (const float*)d_in[1];
    const float* len_w0  = (const float*)d_in[2];
    const float* len_b0  = (const float*)d_in[3];
    const float* len_w1  = (const float*)d_in[4];
    const float* len_b1  = (const float*)d_in[5];
    const float* len_w2  = (const float*)d_in[6];
    const float* len_b2  = (const float*)d_in[7];
    const float* init_w0 = (const float*)d_in[8];
    const float* init_b0 = (const float*)d_in[9];
    const float* init_w1 = (const float*)d_in[10];
    const float* init_b1 = (const float*)d_in[11];
    const float* init_w2 = (const float*)d_in[12];
    const float* init_b2 = (const float*)d_in[13];
    const float* init_w3 = (const float*)d_in[14];
    const float* init_b3 = (const float*)d_in[15];
    const float* rnn_wih = (const float*)d_in[16];
    const float* rnn_whh = (const float*)d_in[17];
    const float* rnn_bih = (const float*)d_in[18];
    const float* rnn_bhh = (const float*)d_in[19];
    const float* fin_w0  = (const float*)d_in[20];
    const float* fin_b0  = (const float*)d_in[21];
    const float* fin_w1  = (const float*)d_in[22];
    const float* fin_b1  = (const float*)d_in[23];
    const float* fin_w2  = (const float*)d_in[24];
    const float* fin_b2  = (const float*)d_in[25];
    const float* fin_w3  = (const float*)d_in[26];
    const float* fin_b3  = (const float*)d_in[27];

    char* ws = (char*)d_ws;
    float*          lv    = (float*)(ws + 0);                 // 1 KB
    float*          hid   = (float*)(ws + 1024);              // 4 KB
    int*            prod  = (int*)(ws + 8192);                // 64 B
    unsigned short* WT0   = (unsigned short*)(ws + 16384);    // 4 x 128 KB
    unsigned short* WT1   = WT0 + C * C;
    unsigned short* WT2   = WT1 + C * C;
    unsigned short* WT3   = WT2 + C * C;
    unsigned short* yA    = (unsigned short*)(ws + (1ull << 20));       // 8 MB
    unsigned short* yB    = yA + (size_t)N_ATOMS * C;                   // 8 MB
    float*          P0    = (float*)(ws + 17825792ull);                 // 16 MB
    float*          Hring = (float*)(ws + 34603008ull);                 // 2 MB

    hipMemsetAsync(prod, 0, 64, stream);

    k_len<<<1, C, 0, stream>>>(l, len_w0, len_b0, len_w1, len_b1, len_w2, len_b2, lv);
    k_in0<<<N_ATOMS, C, 0, stream>>>(x, init_w0, init_b0, yA);

    k_twt<<<C, C, 0, stream>>>(init_w1, WT0);
    k_twt<<<C, C, 0, stream>>>(init_w2, WT1);
    k_twt<<<C, C, 0, stream>>>(init_w3, WT2);
    k_twt<<<C, C, 0, stream>>>(rnn_wih, WT3);   // layer-0 slice

    const size_t ldsBytes = (size_t)C * LDS_STRIDE * sizeof(unsigned short);
    const int    gblocks  = N_ATOMS / 128;
    k_gemm<1, 0><<<gblocks, 256, ldsBytes, stream>>>(yA, WT0, init_b1, yB, nullptr);
    k_gemm<1, 0><<<gblocks, 256, ldsBytes, stream>>>(yB, WT1, init_b2, yA, nullptr);
    k_gemm<0, 0><<<gblocks, 256, ldsBytes, stream>>>(yA, WT2, init_b3, yB, nullptr);
    // P0 = y @ Wih0 + bih0  (f32 out, no activation)
    k_gemm<0, 1><<<gblocks, 256, ldsBytes, stream>>>(yB, WT3, rnn_bih, nullptr, P0);

    k_rnn<<<LAYERS, C, 0, stream>>>(P0, rnn_wih, rnn_whh, rnn_bih, rnn_bhh,
                                    Hring, hid, prod, N_ATOMS);

    k_fin<<<1, C, 0, stream>>>(lv, hid, fin_w0, fin_b0, fin_w1, fin_b1,
                               fin_w2, fin_b2, fin_w3, fin_b3, (float*)d_out);
}
